// RNNLayer_46342697123844
// MI455X (gfx1250) — compile-verified, matrix operand order checked by reading, not run on hardware
//
#include <hip/hip_runtime.h>

#define T_STEPS 512
#define BATCH   64
#define DIM     256
#define HID     256
#define TB      (T_STEPS*BATCH)   // 32768
#define ALPHA_F 0.9f

typedef __attribute__((ext_vector_type(16))) __bf16 v16bf;
typedef __attribute__((ext_vector_type(8)))  float  v8f;

// output offsets (in floats): spikes[T+1,B,H], preds[T,B,H], probs[T,B,H], v_T[B,H], s_T[B,H]
#define OFF_SPIKES 0L
#define OFF_PREDS  ((long)(T_STEPS+1)*BATCH*HID)
#define OFF_PROBS  (OFF_PREDS + (long)T_STEPS*BATCH*HID)
#define OFF_VT     (OFF_PROBS + (long)T_STEPS*BATCH*HID)
#define OFF_ST     (OFF_VT + (long)BATCH*HID)

// K index pattern for 16-bit WMMA A/B fragments (ISA 7.12.2):
// half h of lane -> K = (h&7) + 16*(h>>3) + 8*(lane>>4)
__device__ __forceinline__ int kpat(int h, int laneHi) {
    return (h & 7) + ((h >> 3) << 4) + (laneHi << 3);
}

__device__ __forceinline__ unsigned short f2bfu(float f) {
    __bf16 h = (__bf16)f;
    return __builtin_bit_cast(unsigned short, h);
}

// X workspace is stored fragment-major so each lane owns one contiguous
// 32-byte run per 16x16 tile:  addr(floats) = ((mtile*16 + nb)*32 + lane)*8 + r
// with mtile = t*4 + mb  (global 16-row tile index over TB).

// ---------------------------------------------------------------------------
// Kernel 1: swizzle W_in (fp32 [D,H] row-major) into bf16 WMMA B-fragment
// layout in workspace: frag f = nb*8+kc; element ((f*32+lane)*16+h).
// ---------------------------------------------------------------------------
__global__ void swizzle_win_kernel(const float* __restrict__ W,
                                   unsigned short* __restrict__ dst) {
    int i = blockIdx.x * blockDim.x + threadIdx.x;   // 0..65535
    int f = i >> 9;
    int r = i & 511;
    int lane = r >> 4;
    int h = r & 15;
    int nb = f >> 3, kc = f & 7;
    int K = kc * 32 + kpat(h, lane >> 4);
    int N = nb * 16 + (lane & 15);
    dst[i] = f2bfu(W[K * HID + N]);
}

// ---------------------------------------------------------------------------
// Kernel 2: X = inputs[TB,D] @ W_in[D,H] + b   (bf16 WMMA, f32 accumulate)
// Block = 8 waves; each wave computes one 16x16 C tile; block tile 32x64.
// C written to X in fragment-major layout (2x global_store_b128 per lane).
// ---------------------------------------------------------------------------
__global__ __launch_bounds__(256) void gemm_xin_kernel(
        const float* __restrict__ A,            // [TB, DIM] fp32
        const unsigned short* __restrict__ Bf,  // swizzled W_in bf16 frags
        const float* __restrict__ bias,         // [HID]
        float* __restrict__ X) {                // fragment-major [TB/16*16tiles]
    const int lane   = threadIdx.x & 31;
    const int wave   = threadIdx.x >> 5;        // 0..7
    const int laneHi = lane >> 4;
    const int laneLo = lane & 15;
    const int mtile  = blockIdx.x * 2 + (wave >> 2);   // 0..2047
    const int m0     = mtile * 16;
    const int nb     = blockIdx.y * 4 + (wave & 3);    // global N tile (0..15)
    const int n0     = nb * 16;

    v8f acc;
    const float bv = bias[n0 + laneLo];
#pragma unroll
    for (int r = 0; r < 8; ++r) acc[r] = bv;

    const int row = m0 + laneLo;
    const v16bf* Bfrag = (const v16bf*)Bf;

#pragma unroll
    for (int kc = 0; kc < 8; ++kc) {
        const float* ap = A + (size_t)row * DIM + kc * 32 + (laneHi << 3);
        float4 a0 = *(const float4*)(ap);
        float4 a1 = *(const float4*)(ap + 4);
        float4 a2 = *(const float4*)(ap + 16);
        float4 a3 = *(const float4*)(ap + 20);
        v16bf a;
        a[0]  = (__bf16)a0.x; a[1]  = (__bf16)a0.y; a[2]  = (__bf16)a0.z; a[3]  = (__bf16)a0.w;
        a[4]  = (__bf16)a1.x; a[5]  = (__bf16)a1.y; a[6]  = (__bf16)a1.z; a[7]  = (__bf16)a1.w;
        a[8]  = (__bf16)a2.x; a[9]  = (__bf16)a2.y; a[10] = (__bf16)a2.z; a[11] = (__bf16)a2.w;
        a[12] = (__bf16)a3.x; a[13] = (__bf16)a3.y; a[14] = (__bf16)a3.z; a[15] = (__bf16)a3.w;
        v16bf b = Bfrag[(nb * 8 + kc) * 32 + lane];
        acc = __builtin_amdgcn_wmma_f32_16x16x32_bf16(
                false, a, false, b, (short)0, acc, false, false);
    }
    float* xp = X + ((size_t)(mtile * 16 + nb) * 32 + lane) * 8;
    *(float4*)(xp)     = make_float4(acc[0], acc[1], acc[2], acc[3]);
    *(float4*)(xp + 4) = make_float4(acc[4], acc[5], acc[6], acc[7]);
}

// ---------------------------------------------------------------------------
// Kernel 3: persistent recurrence on one WGP.
// LDS: W_rec frags (128KB) + W_out frags (128KB) + s frags (32KB) = 288KB.
// 16 waves; wave -> (mb = wave&3, nb in {(wave>>2)*4 .. +3}); v in VGPRs.
// ---------------------------------------------------------------------------
__global__ __launch_bounds__(512) void rnn_recurrent_kernel(
        const float* __restrict__ Xin,     // fragment-major x@W_in + b
        const float* __restrict__ spike0,  // [BATCH, HID]
        const float* __restrict__ v0,      // [BATCH, HID]
        const float* __restrict__ Wrec,    // [HID, HID]
        const float* __restrict__ Wout,    // [HID, HID]
        float* __restrict__ out) {
    extern __shared__ char smem[];
    unsigned short* sWrec = (unsigned short*)smem;  // 65536 halfs
    unsigned short* sWout = sWrec + 65536;          // 65536 halfs
    unsigned short* sA    = sWout + 65536;          // 16384 halfs

    const int tid = threadIdx.x;

    // --- one-time: swizzle W_rec / W_out into B-fragment layout ---
    for (int i = tid; i < 65536; i += 512) {
        int f = i >> 9, r = i & 511;
        int lane = r >> 4, h = r & 15;
        int nbF = f >> 3, kcF = f & 7;
        int K = kcF * 32 + kpat(h, lane >> 4);
        int N = nbF * 16 + (lane & 15);
        sWrec[i] = f2bfu(Wrec[K * HID + N]);
        sWout[i] = f2bfu(Wout[K * HID + N]);
    }
    // --- one-time: spike0 into A-fragment layout; also spikes[0] output ---
    for (int i = tid; i < 16384; i += 512) {
        int f = i >> 9, r = i & 511;
        int lane = r >> 4, h = r & 15;
        int mbF = f >> 3, kcF = f & 7;
        int row = mbF * 16 + (lane & 15);
        int col = kcF * 32 + kpat(h, lane >> 4);
        sA[i] = f2bfu(spike0[row * HID + col]);
        out[OFF_SPIKES + i] = spike0[i];
    }

    const int lane   = tid & 31;
    const int wave   = tid >> 5;            // 0..15
    const int mb     = wave & 3;
    const int nbase  = (wave >> 2) * 4;
    const int laneHi = lane >> 4;
    const int laneLo = lane & 15;

    float vacc[4][8];
#pragma unroll
    for (int j = 0; j < 4; ++j) {
        int col = (nbase + j) * 16 + laneLo;
#pragma unroll
        for (int r = 0; r < 8; ++r) {
            int row = mb * 16 + r + (laneHi << 3);
            vacc[j][r] = v0[row * HID + col];
        }
    }
    __syncthreads();

    const v16bf* fWrec = (const v16bf*)sWrec;
    const v16bf* fWout = (const v16bf*)sWout;
    const v16bf* fA    = (const v16bf*)sA;

    v16bf afrag[8];
#pragma unroll
    for (int kc = 0; kc < 8; ++kc) afrag[kc] = fA[(mb * 8 + kc) * 32 + lane];

    float pbuf[4][8];

#pragma unroll 1
    for (int t = 0; t < T_STEPS; ++t) {
        // ---- prefetch next step's X fragments (global_prefetch_b8) ----
        if (t + 1 < T_STEPS) {
#pragma unroll
            for (int j = 0; j < 4; ++j) {
                const float* nx = Xin +
                    ((size_t)(((t + 1) * 4 + mb) * 16 + nbase + j) * 32 + lane) * 8;
                __builtin_prefetch(nx, 0, 1);
            }
        }
        // ---- GEMM1: v = alpha*v + X_t + s@W_rec ; prob = sigmoid(v) ----
#pragma unroll
        for (int j = 0; j < 4; ++j) {
            int nb  = nbase + j;
            int col = nb * 16 + laneLo;
            const float* xp = Xin +
                ((size_t)((t * 4 + mb) * 16 + nb) * 32 + lane) * 8;
            float4 x0 = *(const float4*)(xp);
            float4 x1 = *(const float4*)(xp + 4);
            v8f acc;
            acc[0] = ALPHA_F * vacc[j][0] + x0.x;
            acc[1] = ALPHA_F * vacc[j][1] + x0.y;
            acc[2] = ALPHA_F * vacc[j][2] + x0.z;
            acc[3] = ALPHA_F * vacc[j][3] + x0.w;
            acc[4] = ALPHA_F * vacc[j][4] + x1.x;
            acc[5] = ALPHA_F * vacc[j][5] + x1.y;
            acc[6] = ALPHA_F * vacc[j][6] + x1.z;
            acc[7] = ALPHA_F * vacc[j][7] + x1.w;
#pragma unroll
            for (int kc = 0; kc < 8; ++kc) {
                v16bf b = fWrec[(nb * 8 + kc) * 32 + lane];
                acc = __builtin_amdgcn_wmma_f32_16x16x32_bf16(
                        false, afrag[kc], false, b, (short)0, acc, false, false);
            }
#pragma unroll
            for (int r = 0; r < 8; ++r) {
                int row = mb * 16 + r + (laneHi << 3);
                float v = acc[r];
                vacc[j][r] = v;
                float p = 1.0f / (1.0f + __expf(-v));
                pbuf[j][r] = p;
                long idx = ((long)t * BATCH + row) * HID + col;
                out[OFF_PROBS + idx] = p;
                out[OFF_SPIKES + ((long)(t + 1) * BATCH + row) * HID + col] = p;
                if (t == T_STEPS - 1) out[OFF_ST + (long)row * HID + col] = p;
            }
        }
        __syncthreads();  // all waves past their last read of old sA
        // ---- write new spike into sA in swizzled A-fragment layout ----
#pragma unroll
        for (int j = 0; j < 4; ++j) {
            int nb  = nbase + j;
            int col = nb * 16 + laneLo;
            int kc  = col >> 5;
            int kk  = col & 31;
            int h   = (kk & 7) + ((kk >> 4) << 3);
            int lHi = (kk >> 3) & 1;
#pragma unroll
            for (int r = 0; r < 8; ++r) {
                int rloc  = r + (laneHi << 3);       // row within 16-row tile
                int dlane = rloc + (lHi << 4);
                int idx   = ((mb * 8 + kc) * 32 + dlane) * 16 + h;
                sA[idx] = f2bfu(pbuf[j][r]);
            }
        }
        __syncthreads();  // new s visible
        // ---- reload A fragments (also used by next iteration's GEMM1) ----
#pragma unroll
        for (int kc = 0; kc < 8; ++kc) afrag[kc] = fA[(mb * 8 + kc) * 32 + lane];
        // ---- GEMM2: pred = spike @ W_out ----
#pragma unroll
        for (int j = 0; j < 4; ++j) {
            int nb  = nbase + j;
            int col = nb * 16 + laneLo;
            v8f acc = {0.f, 0.f, 0.f, 0.f, 0.f, 0.f, 0.f, 0.f};
#pragma unroll
            for (int kc = 0; kc < 8; ++kc) {
                v16bf b = fWout[(nb * 8 + kc) * 32 + lane];
                acc = __builtin_amdgcn_wmma_f32_16x16x32_bf16(
                        false, afrag[kc], false, b, (short)0, acc, false, false);
            }
#pragma unroll
            for (int r = 0; r < 8; ++r) {
                int row = mb * 16 + r + (laneHi << 3);
                out[OFF_PREDS + ((long)t * BATCH + row) * HID + col] = acc[r];
            }
        }
    }
    // ---- final membrane potential v_T ----
#pragma unroll
    for (int j = 0; j < 4; ++j) {
        int col = (nbase + j) * 16 + laneLo;
#pragma unroll
        for (int r = 0; r < 8; ++r) {
            int row = mb * 16 + r + (laneHi << 3);
            out[OFF_VT + (long)row * HID + col] = vacc[j][r];
        }
    }
}

// ---------------------------------------------------------------------------
extern "C" void kernel_launch(void* const* d_in, const int* in_sizes, int n_in,
                              void* d_out, int out_size, void* d_ws, size_t ws_size,
                              hipStream_t stream) {
    const float* inputs = (const float*)d_in[0];
    const float* spike0 = (const float*)d_in[1];
    const float* v0     = (const float*)d_in[2];
    const float* W_in   = (const float*)d_in[3];
    const float* W_rec  = (const float*)d_in[4];
    const float* W_out  = (const float*)d_in[5];
    const float* bias   = (const float*)d_in[6];
    float* out = (float*)d_out;

    // workspace: X (fragment-major, TB*HID floats = 32 MB) then swizzled W_in
    float* ws_X = (float*)d_ws;
    unsigned short* ws_win = (unsigned short*)(ws_X + (size_t)TB * HID);

    swizzle_win_kernel<<<256, 256, 0, stream>>>(W_in, ws_win);

    dim3 g(TB / 32, HID / 64);
    gemm_xin_kernel<<<g, 256, 0, stream>>>(inputs, ws_win, bias, ws_X);

    // 288 KB dynamic LDS (<= 320 KB WGP LDS)
    rnn_recurrent_kernel<<<1, 512, 294912, stream>>>(
            ws_X, spike0, v0, W_rec, W_out, out);
}
